// Part_Graph_51539607552575
// MI455X (gfx1250) — matrix operand order repaired and checked by reading, not compiled, not run on hardware
//
#include <hip/hip_runtime.h>
#include <hip/hip_bf16.h>

// ---------------------------------------------------------------------------
// Fused Part-Graph kernel for gfx1250 (MI455X).
//   Heavy GEMM (Wcat[80,256] x p_fea[256, Npix]) via v_wmma_f32_16x16x32_bf16,
//   weights pre-swizzled into WMMA A-fragment order in LDS, p_fea gathered in
//   B-fragment order with packed f32->bf16 converts and NT cache hints.
//   Graph epilogue (att/dp/edges/gate/upd) on VALU, weights broadcast from LDS.
// ---------------------------------------------------------------------------

typedef __attribute__((ext_vector_type(16))) __bf16          v16bf;
typedef __attribute__((ext_vector_type(2)))  __bf16          v2bf;
typedef __attribute__((ext_vector_type(8)))  unsigned int    v8u;
typedef __attribute__((ext_vector_type(8)))  float           v8f;

#define B_SZ    16
#define SPAT    (96 * 96)            // 9216 (H*W)
#define NPIX    (B_SZ * SPAT)        // 147456
#define CIN     256
#define HD      10
#define NPARTS  6
#define MPAD    80                   // 70 outputs padded to 5 tiles of 16
#define WG_PIX  128                  // pixels per workgroup (4 waves x 32)
#define PART_SZ (NPARTS * B_SZ * HD * SPAT)   // 8847360 (one output tensor)

// s_w layout (floats)
#define OFF_WATT   0      // 60
#define OFF_BATT   60     // 6
#define OFF_WDPX   66     // 100
#define OFF_BDP    166    // 10
#define OFF_BPROJ  176    // 60
#define OFF_WGATE  236    // 120
#define OFF_BGATE  356    // 6
#define OFF_WUPD   362    // 1800
#define OFF_BUPD   2162   // 60
#define W_TOTAL    2222

__device__ __forceinline__ unsigned short f32_to_bf16_rne(float f) {
    unsigned int u = __builtin_bit_cast(unsigned int, f);
    u = (u + 0x7fffu + ((u >> 16) & 1u)) >> 16;
    return (unsigned short)u;
}

// Packed f32x2 -> bf16x2: single v_cvt_pk_bf16_f32 when the toolchain has it.
__device__ __forceinline__ unsigned int pack_bf16x2(float lo, float hi) {
#if __has_builtin(__builtin_amdgcn_cvt_pk_bf16_f32)
    v2bf r = __builtin_amdgcn_cvt_pk_bf16_f32(lo, hi);
    return __builtin_bit_cast(unsigned int, r);
#else
    return (unsigned int)f32_to_bf16_rne(lo) |
           ((unsigned int)f32_to_bf16_rne(hi) << 16);
#endif
}

__device__ __forceinline__ float sigmoidf_fast(float x) {
    return 1.0f / (1.0f + __expf(-x));
}

__global__ __launch_bounds__(WG_PIX) void part_graph_fused(
    const float* __restrict__ p_fea,   // [16,256,96,96]
    const float* __restrict__ xp,      // [6,16,10,96,96]
    const float* __restrict__ xh,      // [2,16,10,96,96]
    const float* __restrict__ w_att,   const float* __restrict__ b_att,
    const float* __restrict__ w_dp_f,  const float* __restrict__ w_dp_x,
    const float* __restrict__ b_dp,
    const float* __restrict__ w_proj,  const float* __restrict__ b_proj,
    const float* __restrict__ w_gate,  const float* __restrict__ b_gate,
    const float* __restrict__ w_upd,   const float* __restrict__ b_upd,
    float* __restrict__ out)           // [3 * 8847360]
{
    // 40KB overlay region: bf16 A-fragments during GEMM, then [128][80] f32 lin
    __shared__ __align__(32) unsigned char s_region0[40960];
    __shared__ float s_w[2240];

    const int tid     = threadIdx.x;
    const int lane    = tid & 31;
    const int wv      = tid >> 5;
    const int wg_base = blockIdx.x * WG_PIX;

    // ---- stage small weights into LDS -----------------------------------
    for (int i = tid; i < W_TOTAL; i += WG_PIX) {
        float v;
        if      (i < OFF_BATT)  v = w_att [i];
        else if (i < OFF_WDPX)  v = b_att [i - OFF_BATT];
        else if (i < OFF_BDP)   v = w_dp_x[i - OFF_WDPX];
        else if (i < OFF_BPROJ) v = b_dp  [i - OFF_BDP];
        else if (i < OFF_WGATE) v = b_proj[i - OFF_BPROJ];
        else if (i < OFF_BGATE) v = w_gate[i - OFF_WGATE];
        else if (i < OFF_WUPD)  v = b_gate[i - OFF_BGATE];
        else if (i < OFF_BUPD)  v = w_upd [i - OFF_WUPD];
        else                    v = b_upd [i - OFF_BUPD];
        s_w[i] = v;
    }

    // ---- build Wcat[80,256] bf16 A-fragments in WMMA lane order ---------
    // frag addr: (((kt*5 + t)*32 + lane)*16 + e) ushorts; 32B per lane slice.
    {
        unsigned short* wcat = (unsigned short*)s_region0;
        for (int q = tid; q < 8 * 5 * 32 * 16; q += WG_PIX) {
            int e  = q & 15;
            int l  = (q >> 4) & 31;
            int r  = q >> 9;
            int t  = r % 5;
            int kt = r / 5;
            int m  = t * 16 + (l & 15);
            int hi = l >> 4;
            int v  = e >> 1, hf = e & 1;
            int klocal = hi * 16 + v * 2 + hf;
            int k = kt * 32 + klocal;
            float val = 0.0f;
            if      (m < HD) val = w_dp_f[m * CIN + k];
            else if (m < 70) val = w_proj[(m - HD) * CIN + k];
            wcat[q] = f32_to_bf16_rne(val);
        }
    }
    __syncthreads();

    // ---- WMMA GEMM: lin[80, 32 pixels] per wave -------------------------
    v8f acc[2][5];
    {
        v8f z = {};
        #pragma unroll
        for (int s = 0; s < 2; ++s)
            #pragma unroll
            for (int t = 0; t < 5; ++t) acc[s][t] = z;
    }

    const int g  = lane >> 4;    // B-fragment k-group (0: K 0-15, 1: K 16-31)
    const int nn = lane & 15;    // pixel column within subtile / A row

    const int pix0 = wg_base + wv * 32 + nn;        // subtile 0 pixel
    const int pix1 = pix0 + 16;                     // subtile 1 pixel
    const int b0 = pix0 / SPAT, sp0 = pix0 % SPAT;
    const int b1 = pix1 / SPAT, sp1 = pix1 % SPAT;
    const float* pf0 = p_fea + (size_t)b0 * CIN * SPAT + sp0;
    const float* pf1 = p_fea + (size_t)b1 * CIN * SPAT + sp1;

    const unsigned short* wcat = (const unsigned short*)s_region0;

    for (int kt = 0; kt < 8; ++kt) {
        const int c0 = kt * 32 + g * 16;
        float f0[16], f1[16];
        #pragma unroll
        for (int e = 0; e < 16; ++e) {
            const size_t coff = (size_t)(c0 + e) * SPAT;
            f0[e] = __builtin_nontemporal_load(&pf0[coff]);   // read-once stream
            f1[e] = __builtin_nontemporal_load(&pf1[coff]);
        }
        v8u pu0, pu1;
        #pragma unroll
        for (int e2 = 0; e2 < 8; ++e2) {
            pu0[e2] = pack_bf16x2(f0[2 * e2], f0[2 * e2 + 1]);
            pu1[e2] = pack_bf16x2(f1[2 * e2], f1[2 * e2 + 1]);
        }
        const v16bf bf0 = __builtin_bit_cast(v16bf, pu0);
        const v16bf bf1 = __builtin_bit_cast(v16bf, pu1);
        #pragma unroll
        for (int t = 0; t < 5; ++t) {
            const v16bf af = *(const v16bf*)(wcat + (((kt * 5 + t) * 32 + lane) * 16));
            acc[0][t] = __builtin_amdgcn_wmma_f32_16x16x32_bf16(
                false, af, false, bf0, (short)0, acc[0][t], false, false);
            acc[1][t] = __builtin_amdgcn_wmma_f32_16x16x32_bf16(
                false, af, false, bf1, (short)0, acc[1][t], false, false);
        }
    }
    __syncthreads();   // all waves done with wcat -> overlay becomes lin

    // ---- scatter accumulators to lin[128][80] ---------------------------
    float* lin = (float*)s_region0;
    #pragma unroll
    for (int s = 0; s < 2; ++s)
        #pragma unroll
        for (int t = 0; t < 5; ++t)
            #pragma unroll
            for (int j = 0; j < 8; ++j)
                lin[(wv * 32 + s * 16 + nn) * MPAD + t * 16 + j + 8 * g] =
                    acc[s][t][j];
    __syncthreads();

    // ---- graph epilogue: one pixel per thread ---------------------------
    const int pix = wg_base + tid;
    const int b   = pix / SPAT, sp = pix % SPAT;
    const float* linme = lin + tid * MPAD;

    float xpv[NPARTS][HD];
    #pragma unroll
    for (int p = 0; p < NPARTS; ++p)
        #pragma unroll
        for (int d = 0; d < HD; ++d)
            xpv[p][d] = __builtin_nontemporal_load(
                &xp[(size_t)((p * B_SZ + b) * HD + d) * SPAT + sp]);

    float xhv[2][HD];
    #pragma unroll
    for (int hf = 0; hf < 2; ++hf)
        #pragma unroll
        for (int d = 0; d < HD; ++d)
            xhv[hf][d] = __builtin_nontemporal_load(
                &xh[(size_t)((hf * B_SZ + b) * HD + d) * SPAT + sp]);

    // attention per part
    float att[NPARTS];
    #pragma unroll
    for (int p = 0; p < NPARTS; ++p) {
        float a = s_w[OFF_BATT + p];
        #pragma unroll
        for (int d = 0; d < HD; ++d) a += s_w[OFF_WATT + p * HD + d] * xpv[p][d];
        att[p] = sigmoidf_fast(a);
    }

    // shared dp_f projection from the GEMM
    float linA[HD];
    #pragma unroll
    for (int d = 0; d < HD; ++d) linA[d] = linme[d];

    // dp per part, reduced on the fly over the star graph centered on part 1
    float dp1[HD], S1[HD];
    #pragma unroll
    for (int d = 0; d < HD; ++d) S1[d] = 0.0f;
    float Ssum = 0.0f;
    #pragma unroll
    for (int p = 0; p < NPARTS; ++p) {
        #pragma unroll
        for (int d = 0; d < HD; ++d) {
            float v = linA[d] + s_w[OFF_BDP + d];
            #pragma unroll
            for (int c = 0; c < HD; ++c)
                v += s_w[OFF_WDPX + d * HD + c] * xpv[p][c];
            v = fmaxf(v, 0.0f);
            if (p == 1) dp1[d] = v;
            else        S1[d] += att[p] * v;
        }
        if (p != 1) Ssum += att[p];
    }

    // per-part: xpp (edge aggregation), xhp (gated projection), upd, stores
    #pragma unroll
    for (int p = 0; p < NPARTS; ++p) {
        float xpp[HD], xhp[HD];
        if (p == 1) {
            #pragma unroll
            for (int d = 0; d < HD; ++d)
                xpp[d] = att[1] * (S1[d] + xpv[1][d] * Ssum);
        } else {
            const float aa = att[1] * att[p];
            #pragma unroll
            for (int d = 0; d < HD; ++d)
                xpp[d] = aa * (dp1[d] + xpv[p][d]);
        }

        const int hf = (p >= 4) ? 1 : 0;
        float ga = s_w[OFF_BGATE + p];
        #pragma unroll
        for (int d = 0; d < HD; ++d)
            ga += s_w[OFF_WGATE + p * 20 + d] * xhv[hf][d]
                + s_w[OFF_WGATE + p * 20 + HD + d] * xpv[p][d];
        const float gate = sigmoidf_fast(ga);
        #pragma unroll
        for (int d = 0; d < HD; ++d)
            xhp[d] = gate * (linme[HD + p * HD + d] + s_w[OFF_BPROJ + p * HD + d]);

        #pragma unroll
        for (int d = 0; d < HD; ++d) {
            float u = s_w[OFF_BUPD + p * HD + d];
            const int wb = OFF_WUPD + (p * HD + d) * 30;
            #pragma unroll
            for (int c = 0; c < HD; ++c) u += s_w[wb + c]          * xpv[p][c];
            #pragma unroll
            for (int c = 0; c < HD; ++c) u += s_w[wb + HD + c]     * xpp[c];
            #pragma unroll
            for (int c = 0; c < HD; ++c) u += s_w[wb + 2 * HD + c] * xhp[c];
            u = fmaxf(u, 0.0f);
            const size_t oi = (size_t)((p * B_SZ + b) * HD + d) * SPAT + sp;
            __builtin_nontemporal_store(xpv[p][d] + u, &out[oi]);              // xp_out
            __builtin_nontemporal_store(xpp[d], &out[(size_t)PART_SZ + oi]);   // xpp
            __builtin_nontemporal_store(xhp[d], &out[2 * (size_t)PART_SZ + oi]); // xhp
        }
    }
}

extern "C" void kernel_launch(void* const* d_in, const int* in_sizes, int n_in,
                              void* d_out, int out_size, void* d_ws, size_t ws_size,
                              hipStream_t stream) {
    (void)in_sizes; (void)n_in; (void)out_size; (void)d_ws; (void)ws_size;
    const float* p_fea  = (const float*)d_in[0];
    const float* xp     = (const float*)d_in[1];
    const float* xh     = (const float*)d_in[2];
    const float* w_att  = (const float*)d_in[3];
    const float* b_att  = (const float*)d_in[4];
    const float* w_dp_f = (const float*)d_in[5];
    const float* w_dp_x = (const float*)d_in[6];
    const float* b_dp   = (const float*)d_in[7];
    const float* w_proj = (const float*)d_in[8];
    const float* b_proj = (const float*)d_in[9];
    const float* w_gate = (const float*)d_in[10];
    const float* b_gate = (const float*)d_in[11];
    const float* w_upd  = (const float*)d_in[12];
    const float* b_upd  = (const float*)d_in[13];
    float* out = (float*)d_out;

    dim3 grid(NPIX / WG_PIX);   // 1152 workgroups, exact tiling
    dim3 block(WG_PIX);         // 4 waves of 32
    part_graph_fused<<<grid, block, 0, stream>>>(
        p_fea, xp, xh, w_att, b_att, w_dp_f, w_dp_x, b_dp,
        w_proj, b_proj, w_gate, b_gate, w_upd, b_upd, out);
}
